// Self_Attention_35536559407619
// MI455X (gfx1250) — matrix operand order repaired and checked by reading, not compiled, not run on hardware
//
#include <hip/hip_runtime.h>
#include <hip/hip_bf16.h>

// ---------------- problem constants ----------------
constexpr int B = 8, S = 1024, E = 1024, H = 8, D = 128;
constexpr int HD = H * D;          // 1024
constexpr int M_TOT = B * S;       // 8192 rows

typedef __bf16 bf16_t;
typedef __attribute__((ext_vector_type(16))) __bf16 v16bf;
typedef __attribute__((ext_vector_type(8)))  __bf16 v8bf;
typedef __attribute__((ext_vector_type(8)))  float  v8f;

// sched_group_barrier masks
#define SG_WMMA     0x008
#define SG_VMEM_RD  0x020
#define SG_DS_RD    0x100

// ---------------- WMMA helpers (CDNA5 wave32, 16x16x32 bf16) ----------------
__device__ __forceinline__ v8f wmma_bf16(v16bf a, v16bf b, v8f c) {
  // 8 args: (neg_a, A, neg_b, B, c_mod, C, reuse_a, reuse_b)
  return __builtin_amdgcn_wmma_f32_16x16x32_bf16(false, a, false, b, (short)0, c,
                                                 false, false);
}

// A operand 16x32: lane = half*16 + m; element i<8 -> K = half*8 + i,
// element i>=8 -> K = half*8 + 16 + (i-8).  Two 16B contiguous runs per lane.
__device__ __forceinline__ v16bf load_A_bf16(const bf16_t* p0, int ld, int k0) {
  const int lane = threadIdx.x & 31;
  const int m = lane & 15, half = lane >> 4;
  const bf16_t* p = p0 + (size_t)m * ld + k0 + half * 8;
  v8bf lo = *(const v8bf*)p;         // K = half*8 + 0..7
  v8bf hi = *(const v8bf*)(p + 16);  // K = half*8 + 16..23
  return __builtin_shufflevector(lo, hi, 0,1,2,3,4,5,6,7,8,9,10,11,12,13,14,15);
}

// Same A layout, converting from fp32 memory on the fly.
__device__ __forceinline__ v16bf load_A_f32(const float* p0, int ld, int k0) {
  const int lane = threadIdx.x & 31;
  const int m = lane & 15, half = lane >> 4;
  const float* p = p0 + (size_t)m * ld + k0 + half * 8;
  v16bf a;
#pragma unroll
  for (int i = 0; i < 8; ++i) { a[i] = (bf16_t)p[i]; a[i + 8] = (bf16_t)p[16 + i]; }
  return a;
}

// B operand 32x16: lane = half*16 + n; element i<8 -> K = half*8 + i,
// element i>=8 -> K = half*8 + 16 + (i-8).  Two 16B contiguous runs per lane.
__device__ __forceinline__ v16bf load_B_bf16(const bf16_t* p0, int ldn, int k0) {
  const int lane = threadIdx.x & 31;
  const int n = lane & 15, half = lane >> 4;
  const bf16_t* p = p0 + (size_t)n * ldn + k0 + half * 8;
  v8bf lo = *(const v8bf*)p;
  v8bf hi = *(const v8bf*)(p + 16);
  return __builtin_shufflevector(lo, hi, 0,1,2,3,4,5,6,7,8,9,10,11,12,13,14,15);
}

// ---------------- prep: weight conversions ----------------
// w [H,E,D] fp32 -> wT [H,D,E] bf16   (B operand of the head projections)
__global__ void convert_wT_kernel(const float* __restrict__ w, bf16_t* __restrict__ wT) {
  size_t idx = (size_t)blockIdx.x * blockDim.x + threadIdx.x;
  if (idx >= (size_t)H * E * D) return;
  int d = (int)(idx % D);
  size_t t = idx / D;
  int e = (int)(t % E);
  int h = (int)(t / E);
  wT[((size_t)h * D + d) * E + e] = (bf16_t)w[idx];
}

// straight fp32 -> bf16 (proj_w keeps its natural [E, H*D] layout)
__global__ void convert_bf_kernel(const float* __restrict__ src, bf16_t* __restrict__ dst,
                                  int n) {
  int idx = blockIdx.x * blockDim.x + threadIdx.x;
  if (idx < n) dst[idx] = (bf16_t)src[idx];
}

// ---------------- stage 1: per-head projection  Y = tanh(X * W) ----------------
// X: [M_TOT, E] fp32.  WT: [H, D, E] bf16.
// mode 0 (k): write Y to nat [H, M, D] and transposed t [H, B, D, S].
// mode 1 (q): write nat only; if *flagp != 0, bypass (qx = q broadcast).
__global__ __launch_bounds__(256) void proj_kernel(
    const float* __restrict__ X, const bf16_t* __restrict__ WT,
    bf16_t* __restrict__ out_nat, bf16_t* __restrict__ out_t,
    const int* __restrict__ flagp, int mode) {
  const int h = blockIdx.y;
  const int w = threadIdx.x >> 5;
  const int lane = threadIdx.x & 31;
  const int n_in = lane & 15, half = lane >> 4;
  const int mrow = blockIdx.x * 128 + w * 16;

  if (mode == 1 && *flagp != 0) {  // q already in hidden space (valid when E==D)
    for (int idx = lane; idx < 16 * D; idx += 32) {
      int r = idx >> 7, d = idx & (D - 1);
      out_nat[((size_t)h * M_TOT + mrow + r) * D + d] = (bf16_t)X[(size_t)(mrow + r) * E + d];
    }
    return;
  }

  v8f acc[8];
  v8f zero = {};
#pragma unroll
  for (int nt = 0; nt < 8; ++nt) acc[nt] = zero;

  const float* arow = X + (size_t)mrow * E;
  const bf16_t* wbase = WT + (size_t)h * D * E;

  // -------- ping-pong pipelined K loop: loads for k+32 overlap WMMAs for k ----
  v16bf a0, a1;
  v16bf b0[8], b1[8];
  a0 = load_A_f32(arow, E, 0);
#pragma unroll
  for (int nt = 0; nt < 8; ++nt) b0[nt] = load_B_bf16(wbase + (size_t)(nt * 16) * E, E, 0);

  int k0 = 0;
  for (; k0 + 64 < E; k0 += 64) {
    a1 = load_A_f32(arow, E, k0 + 32);
#pragma unroll
    for (int nt = 0; nt < 8; ++nt)
      b1[nt] = load_B_bf16(wbase + (size_t)(nt * 16) * E, E, k0 + 32);
#pragma unroll
    for (int nt = 0; nt < 8; ++nt) acc[nt] = wmma_bf16(a0, b0[nt], acc[nt]);
    __builtin_amdgcn_sched_group_barrier(SG_VMEM_RD, 20, 0);
    __builtin_amdgcn_sched_group_barrier(SG_WMMA, 8, 0);

    a0 = load_A_f32(arow, E, k0 + 64);
#pragma unroll
    for (int nt = 0; nt < 8; ++nt)
      b0[nt] = load_B_bf16(wbase + (size_t)(nt * 16) * E, E, k0 + 64);
#pragma unroll
    for (int nt = 0; nt < 8; ++nt) acc[nt] = wmma_bf16(a1, b1[nt], acc[nt]);
    __builtin_amdgcn_sched_group_barrier(SG_VMEM_RD, 20, 0);
    __builtin_amdgcn_sched_group_barrier(SG_WMMA, 8, 0);
  }
  // epilogue: k0 == E-64; step E-64 is in (a0,b0), load final step E-32
  a1 = load_A_f32(arow, E, E - 32);
#pragma unroll
  for (int nt = 0; nt < 8; ++nt)
    b1[nt] = load_B_bf16(wbase + (size_t)(nt * 16) * E, E, E - 32);
#pragma unroll
  for (int nt = 0; nt < 8; ++nt) acc[nt] = wmma_bf16(a0, b0[nt], acc[nt]);
#pragma unroll
  for (int nt = 0; nt < 8; ++nt) acc[nt] = wmma_bf16(a1, b1[nt], acc[nt]);

#pragma unroll
  for (int nt = 0; nt < 8; ++nt) {
#pragma unroll
    for (int r = 0; r < 8; ++r) {
      float v = tanhf(acc[nt][r]);
      int m = mrow + r + 8 * half;
      int d = nt * 16 + n_in;
      out_nat[((size_t)h * M_TOT + m) * D + d] = (bf16_t)v;
      if (mode == 0) {
        int b = m >> 10, s = m & (S - 1);
        out_t[(((size_t)h * B + b) * D + d) * S + s] = (bf16_t)v;
      }
    }
  }
}

// ---------------- stage 2+3: flash attention per (h, b, q-tile) ----------------
// qx:[H, B*S, D] bf16, kx_nat:[H, B*S, D] bf16, kx_t:[H,B,D,S] bf16
// out: attn_out [B, S, H*D] bf16 (heads concatenated, ready as A for stage 4)
__global__ __launch_bounds__(256) void attn_kernel(
    const bf16_t* __restrict__ qx, const bf16_t* __restrict__ kx_nat,
    const bf16_t* __restrict__ kx_t, bf16_t* __restrict__ attn_out) {
  const int b = blockIdx.y, h = blockIdx.z;
  const int w = threadIdx.x >> 5;
  const int lane = threadIdx.x & 31;
  const int n_in = lane & 15, half = lane >> 4;
  const int q0 = blockIdx.x * 128 + w * 16;

  __shared__ bf16_t ldsP[8 * 16 * 128];          // 32 KB, 4 KB per wave
  bf16_t* myP = ldsP + w * (16 * 128);

  const size_t hb = (size_t)h * B + b;
  const bf16_t* qbase = qx + (hb * S + q0) * D;
  const bf16_t* knat = kx_nat + hb * S * D;
  const bf16_t* kt = kx_t + hb * (size_t)D * S;

  v16bf Aq[4];
#pragma unroll
  for (int kk = 0; kk < 4; ++kk) Aq[kk] = load_A_bf16(qbase, D, kk * 32);

  v8f O[8];
  v8f zero = {};
  float rmax[8], rsum[8];
#pragma unroll
  for (int nt = 0; nt < 8; ++nt) O[nt] = zero;
#pragma unroll
  for (int r = 0; r < 8; ++r) { rmax[r] = -3.0e38f; rsum[r] = 0.0f; }

  const float scale = 0.088388347648318447f;  // 1/sqrt(128)

  for (int j = 0; j < S / 128; ++j) {
    // ---- scores: S = qx * kx^T  (B operand from kx_nat: K=d contiguous) ----
    v8f Sc[8];
#pragma unroll
    for (int nt = 0; nt < 8; ++nt) Sc[nt] = zero;

    v16bf bs_cur[8];
#pragma unroll
    for (int nt = 0; nt < 8; ++nt)
      bs_cur[nt] = load_B_bf16(knat + (size_t)(j * 128 + nt * 16) * D, D, 0);
#pragma unroll
    for (int kk = 0; kk < 4; ++kk) {   // fully unrolled: copies become SSA renames
      v16bf bs_nxt[8];
      if (kk < 3) {
#pragma unroll
        for (int nt = 0; nt < 8; ++nt)
          bs_nxt[nt] = load_B_bf16(knat + (size_t)(j * 128 + nt * 16) * D, D, (kk + 1) * 32);
      }
#pragma unroll
      for (int nt = 0; nt < 8; ++nt)
        Sc[nt] = wmma_bf16(Aq[kk], bs_cur[nt], Sc[nt]);
      if (kk < 3) {
        __builtin_amdgcn_sched_group_barrier(SG_VMEM_RD, 16, 0);
        __builtin_amdgcn_sched_group_barrier(SG_WMMA, 8, 0);
#pragma unroll
        for (int nt = 0; nt < 8; ++nt) bs_cur[nt] = bs_nxt[nt];
      }
    }

    // ---- online softmax (row m = r + 8*half lives in one 16-lane half) ----
#pragma unroll
    for (int r = 0; r < 8; ++r) {
      float mloc = -3.0e38f;
#pragma unroll
      for (int nt = 0; nt < 8; ++nt) {
        float v = Sc[nt][r] * scale;
        Sc[nt][r] = v;
        mloc = fmaxf(mloc, v);
      }
#pragma unroll
      for (int off = 1; off <= 8; off <<= 1) mloc = fmaxf(mloc, __shfl_xor(mloc, off, 32));
      float mnew = fmaxf(rmax[r], mloc);
      float corr = __expf(rmax[r] - mnew);
      rmax[r] = mnew;
      float ps = 0.0f;
#pragma unroll
      for (int nt = 0; nt < 8; ++nt) {
        float e = __expf(Sc[nt][r] - mnew);
        Sc[nt][r] = e;
        ps += e;
      }
#pragma unroll
      for (int off = 1; off <= 8; off <<= 1) ps += __shfl_xor(ps, off, 32);
      rsum[r] = rsum[r] * corr + ps;
#pragma unroll
      for (int nt = 0; nt < 8; ++nt) O[nt][r] *= corr;
    }

    // ---- P relayout through per-wave LDS (C layout -> A layout) ----
#pragma unroll
    for (int nt = 0; nt < 8; ++nt)
#pragma unroll
      for (int r = 0; r < 8; ++r)
        myP[(r + 8 * half) * 128 + nt * 16 + n_in] = (bf16_t)Sc[nt][r];

    // ---- O += P * kx  (B operand from kx_t: K=s contiguous) ----
    v16bf Ap_cur = load_A_bf16(myP, 128, 0);
    v16bf bo_cur[8];
#pragma unroll
    for (int nt = 0; nt < 8; ++nt)
      bo_cur[nt] = load_B_bf16(kt + (size_t)(nt * 16) * S, S, j * 128);
#pragma unroll
    for (int kk = 0; kk < 4; ++kk) {   // fully unrolled: copies become SSA renames
      v16bf Ap_nxt;
      v16bf bo_nxt[8];
      if (kk < 3) {
        Ap_nxt = load_A_bf16(myP, 128, (kk + 1) * 32);
#pragma unroll
        for (int nt = 0; nt < 8; ++nt)
          bo_nxt[nt] = load_B_bf16(kt + (size_t)(nt * 16) * S, S, j * 128 + (kk + 1) * 32);
      }
#pragma unroll
      for (int nt = 0; nt < 8; ++nt)
        O[nt] = wmma_bf16(Ap_cur, bo_cur[nt], O[nt]);
      if (kk < 3) {
        __builtin_amdgcn_sched_group_barrier(SG_DS_RD, 2, 0);
        __builtin_amdgcn_sched_group_barrier(SG_VMEM_RD, 16, 0);
        __builtin_amdgcn_sched_group_barrier(SG_WMMA, 8, 0);
        Ap_cur = Ap_nxt;
#pragma unroll
        for (int nt = 0; nt < 8; ++nt) bo_cur[nt] = bo_nxt[nt];
      }
    }
  }

  // ---- finalize: divide by row sums, write concat-head layout ----
#pragma unroll
  for (int r = 0; r < 8; ++r) {
    float inv = 1.0f / rsum[r];
    int m = q0 + r + 8 * half;
#pragma unroll
    for (int nt = 0; nt < 8; ++nt) {
      attn_out[((size_t)b * S + m) * HD + h * D + nt * 16 + n_in] =
          (bf16_t)(O[nt][r] * inv);
    }
  }
}

// ---------------- stage 4: output projection  out = AO * proj_w^T + b ----------------
// AO:[M_TOT, HD] bf16, Wb = proj_w bf16 [E, HD] (natural layout == B operand layout)
__global__ __launch_bounds__(256) void outproj_kernel(
    const bf16_t* __restrict__ AO, const bf16_t* __restrict__ Wb,
    const float* __restrict__ bias, float* __restrict__ out) {
  const int n0 = blockIdx.y * 128;
  const int w = threadIdx.x >> 5;
  const int lane = threadIdx.x & 31;
  const int n_in = lane & 15, half = lane >> 4;
  const int mrow = blockIdx.x * 128 + w * 16;

  v8f acc[8];
  v8f zero = {};
#pragma unroll
  for (int nt = 0; nt < 8; ++nt) acc[nt] = zero;

  const bf16_t* arow = AO + (size_t)mrow * HD;

  // -------- ping-pong pipelined K loop ----
  v16bf a0, a1;
  v16bf b0[8], b1[8];
  a0 = load_A_bf16(arow, HD, 0);
#pragma unroll
  for (int nt = 0; nt < 8; ++nt)
    b0[nt] = load_B_bf16(Wb + (size_t)(n0 + nt * 16) * HD, HD, 0);

  int k0 = 0;
  for (; k0 + 64 < HD; k0 += 64) {
    a1 = load_A_bf16(arow, HD, k0 + 32);
#pragma unroll
    for (int nt = 0; nt < 8; ++nt)
      b1[nt] = load_B_bf16(Wb + (size_t)(n0 + nt * 16) * HD, HD, k0 + 32);
#pragma unroll
    for (int nt = 0; nt < 8; ++nt) acc[nt] = wmma_bf16(a0, b0[nt], acc[nt]);
    __builtin_amdgcn_sched_group_barrier(SG_VMEM_RD, 18, 0);
    __builtin_amdgcn_sched_group_barrier(SG_WMMA, 8, 0);

    a0 = load_A_bf16(arow, HD, k0 + 64);
#pragma unroll
    for (int nt = 0; nt < 8; ++nt)
      b0[nt] = load_B_bf16(Wb + (size_t)(n0 + nt * 16) * HD, HD, k0 + 64);
#pragma unroll
    for (int nt = 0; nt < 8; ++nt) acc[nt] = wmma_bf16(a1, b1[nt], acc[nt]);
    __builtin_amdgcn_sched_group_barrier(SG_VMEM_RD, 18, 0);
    __builtin_amdgcn_sched_group_barrier(SG_WMMA, 8, 0);
  }
  // epilogue: step HD-64 in (a0,b0); load final step HD-32
  a1 = load_A_bf16(arow, HD, HD - 32);
#pragma unroll
  for (int nt = 0; nt < 8; ++nt)
    b1[nt] = load_B_bf16(Wb + (size_t)(n0 + nt * 16) * HD, HD, HD - 32);
#pragma unroll
  for (int nt = 0; nt < 8; ++nt) acc[nt] = wmma_bf16(a0, b0[nt], acc[nt]);
#pragma unroll
  for (int nt = 0; nt < 8; ++nt) acc[nt] = wmma_bf16(a1, b1[nt], acc[nt]);

#pragma unroll
  for (int nt = 0; nt < 8; ++nt) {
    int e = n0 + nt * 16 + n_in;
    float bb = bias[e];
#pragma unroll
    for (int r = 0; r < 8; ++r)
      out[(size_t)(mrow + r + 8 * half) * E + e] = acc[nt][r] + bb;
  }
}

// ---------------- launcher ----------------
extern "C" void kernel_launch(void* const* d_in, const int* in_sizes, int n_in,
                              void* d_out, int out_size, void* d_ws, size_t ws_size,
                              hipStream_t stream) {
  const float* k_in  = (const float*)d_in[0];
  const float* q_in  = (const float*)d_in[1];
  const int*   flagp = (const int*)d_in[2];
  const float* w_kx  = (const float*)d_in[3];
  const float* w_qx  = (const float*)d_in[4];
  const float* projw = (const float*)d_in[5];
  const float* projb = (const float*)d_in[6];
  float* out = (float*)d_out;

  // workspace carve-up (bf16 elements)
  constexpr size_t SZ_WT = (size_t)H * D * E;     // 1 Mi elems each
  constexpr size_t SZ_PW = (size_t)E * HD;        // 1 Mi elems
  constexpr size_t SZ_X  = (size_t)H * M_TOT * D; // 8 Mi elems each
  constexpr size_t SZ_AO = (size_t)M_TOT * HD;    // 8 Mi elems

  char* ws = (char*)d_ws;
  bf16_t* wkxT = (bf16_t*)ws;                          ws += SZ_WT * 2;
  bf16_t* wqxT = (bf16_t*)ws;                          ws += SZ_WT * 2;
  bf16_t* pwbf = (bf16_t*)ws;                          ws += SZ_PW * 2;
  bf16_t* qx   = (bf16_t*)ws;                          ws += SZ_X * 2;
  bf16_t* kxn  = (bf16_t*)ws;                          ws += SZ_X * 2;
  bf16_t* kxt  = (bf16_t*)ws;                          ws += SZ_X * 2;
  bf16_t* ao   = (bf16_t*)ws;                          ws += SZ_AO * 2;
  (void)ws_size; (void)in_sizes; (void)n_in; (void)out_size;

  const int cvt_blocks = (int)((SZ_WT + 255) / 256);
  convert_wT_kernel<<<cvt_blocks, 256, 0, stream>>>(w_kx, wkxT);
  convert_wT_kernel<<<cvt_blocks, 256, 0, stream>>>(w_qx, wqxT);
  convert_bf_kernel<<<(int)((SZ_PW + 255) / 256), 256, 0, stream>>>(projw, pwbf, (int)SZ_PW);

  // head projections: grid = (M tiles of 128, H), 8 waves per block
  proj_kernel<<<dim3(M_TOT / 128, H), 256, 0, stream>>>(k_in, wkxT, kxn, kxt, flagp, 0);
  proj_kernel<<<dim3(M_TOT / 128, H), 256, 0, stream>>>(q_in, wqxT, qx,  qx,  flagp, 1);

  // flash attention: grid = (Sq tiles, B, H)
  attn_kernel<<<dim3(S / 128, B, H), 256, 0, stream>>>(qx, kxn, kxt, ao);

  // output projection: grid = (M tiles, N tiles)
  outproj_kernel<<<dim3(M_TOT / 128, E / 128), 256, 0, stream>>>(ao, pwbf, projb, out);
}
